// MultiheadAttentionWithRoPE_7876970021197
// MI455X (gfx1250) — matrix operand order repaired and checked by reading, not compiled, not run on hardware
//
#include <hip/hip_runtime.h>
#include <cstdint>
#include <cstddef>

// ---------------------------------------------------------------------------
// MultiheadAttention + RoPE for MI455X (gfx1250), wave32 + WMMA f16->f32.
// N=2048, L=50, E=512, H=8, HD=64.  All GEMMs use v_wmma_f32_16x16x32_f16.
// ---------------------------------------------------------------------------

typedef __attribute__((ext_vector_type(16))) _Float16 v16h;
typedef __attribute__((ext_vector_type(8)))  _Float16 v8h;
typedef __attribute__((ext_vector_type(4)))  _Float16 v4h;
typedef __attribute__((ext_vector_type(8)))  float    v8f;

#define DEVINL __device__ __forceinline__

DEVINL v8f wmma16(v16h a, v16h b, v8f c) {
  // D = A(16x32 f16) * B(32x16 f16) + C(16x16 f32)
  return __builtin_amdgcn_wmma_f32_16x16x32_f16(
      /*neg_a=*/false, a, /*neg_b=*/false, b,
      /*c_mod=*/(short)0, c, /*reuse_a=*/false, /*reuse_b=*/false);
}

// A fragment (16x32 f16), buf is row-major [M][stride] in LDS.
// ISA 7.12.2: lane = row (l&15); halves 0..7 -> K = g*8+r ; halves 8..15 -> K = 16+g*8+r
DEVINL v16h ldfragA(const _Float16* buf, int r0, int stride, int k0) {
  const int l = threadIdx.x & 31;
  const int r = r0 + (l & 15);
  const int g = (l >> 4) << 3;               // 0 or 8
  const v8h lo = *(const v8h*)(buf + r * stride + k0 + g);
  const v8h hi = *(const v8h*)(buf + r * stride + k0 + 16 + g);
  v16h o;
#pragma unroll
  for (int i = 0; i < 8; ++i) { o[i] = lo[i]; o[i + 8] = hi[i]; }
  return o;
}

// B fragment (32x16 f16) loaded from row-major [N][stride] buffer (B[k][n] = buf[n][k]).
// lane = column n (l&15); halves 0..7 -> K = g*8+r ; halves 8..15 -> K = 16+g*8+r (same as A)
DEVINL v16h ldfragB(const _Float16* buf, int n0, int stride, int k0) {
  const int l = threadIdx.x & 31;
  const int r = n0 + (l & 15);
  const int g = (l >> 4) << 3;               // 0 or 8
  const v8h lo = *(const v8h*)(buf + r * stride + k0 + g);
  const v8h hi = *(const v8h*)(buf + r * stride + k0 + 16 + g);
  v16h o;
#pragma unroll
  for (int i = 0; i < 8; ++i) { o[i] = lo[i]; o[i + 8] = hi[i]; }
  return o;
}

// ---------------------------------------------------------------------------
// Generic GEMM:  C[M x Ncall] = A[M x K] * W[Ncall x K]^T + bias
// Block tile 128(M) x 128(N), K-stage 64; 256 threads = 8 waves, each wave a
// 32x64 subtile (2x4 WMMA tiles, 16 WMMAs per barrier pair).
// Global loads for the next stage issue before the barrier -> overlap compute.
// A is f32 (converted to f16 in LDS) or f16; C written as f16 or f32.
// ---------------------------------------------------------------------------
template <bool AHALF, bool CHALF>
__global__ __launch_bounds__(256) void gemm_wmma_kernel(
    const void* __restrict__ Av, const float* __restrict__ W,
    const float* __restrict__ bias, void* __restrict__ Cv,
    int K, int ldc) {
  __shared__ _Float16 lA[128 * 80];   // 20480 B, row stride 80 halves (160 B)
  __shared__ _Float16 lB[128 * 80];   // 20480 B

  const int m0 = blockIdx.y * 128;
  const int n0 = blockIdx.x * 128;
  const int t = threadIdx.x;
  const int w = t >> 5;
  const int lane = t & 31;
  const int wm = w & 3;                // 4 waves along M (32 rows each)
  const int wn = w >> 2;               // 2 waves along N (64 cols each)

  v8f acc[2][4] = {};

  for (int k0 = 0; k0 < K; k0 += 64) {
    // ---- issue global loads for this stage into registers ------------------
    v8h  ha[4];                        // AHALF path: 4 chunks of 8 halves
    float4 ra[8];                      // f32 path:   8 chunks of float4
    float4 rw[8];                      // W tile:     8 chunks of float4
    if (AHALF) {
      const _Float16* A = (const _Float16*)Av;
#pragma unroll
      for (int j = 0; j < 4; ++j) {
        const int c = t + 256 * j;                 // 1024 chunks of 8 halves
        const int row = c >> 3;
        const int kc = (c & 7) << 3;
        ha[j] = *(const v8h*)(A + (size_t)(m0 + row) * K + k0 + kc);
      }
    } else {
      const float* A = (const float*)Av;
#pragma unroll
      for (int j = 0; j < 8; ++j) {
        const int c = t + 256 * j;                 // 2048 chunks of float4
        const int row = c >> 4;
        const int kc = (c & 15) << 2;
        ra[j] = *(const float4*)(A + (size_t)(m0 + row) * K + k0 + kc);
      }
    }
#pragma unroll
    for (int j = 0; j < 8; ++j) {
      const int c = t + 256 * j;                   // 2048 chunks of float4
      const int row = c >> 4;
      const int kc = (c & 15) << 2;
      rw[j] = *(const float4*)(W + (size_t)(n0 + row) * K + k0 + kc);
    }
    // prefetch next K-stage into GL2 (gfx1250 global_prefetch_b8)
    if (k0 + 64 < K) {
      if (AHALF) {
        const _Float16* A = (const _Float16*)Av;
        __builtin_prefetch(A + (size_t)(m0 + (t >> 1)) * K + k0 + 64, 0, 1);
      } else {
        const float* A = (const float*)Av;
        __builtin_prefetch(A + (size_t)(m0 + (t >> 1)) * K + k0 + 64, 0, 1);
      }
      __builtin_prefetch(W + (size_t)(n0 + (t >> 1)) * K + k0 + 64, 0, 1);
    }

    __syncthreads();   // previous stage's compute done with LDS

    // ---- store stage to LDS (convert f32 -> f16 where needed) --------------
    if (AHALF) {
#pragma unroll
      for (int j = 0; j < 4; ++j) {
        const int c = t + 256 * j;
        const int row = c >> 3;
        const int kc = (c & 7) << 3;
        *(v8h*)(lA + row * 80 + kc) = ha[j];
      }
    } else {
#pragma unroll
      for (int j = 0; j < 8; ++j) {
        const int c = t + 256 * j;
        const int row = c >> 4;
        const int kc = (c & 15) << 2;
        v4h hv;
        hv[0] = (_Float16)ra[j].x; hv[1] = (_Float16)ra[j].y;
        hv[2] = (_Float16)ra[j].z; hv[3] = (_Float16)ra[j].w;
        *(v4h*)(lA + row * 80 + kc) = hv;
      }
    }
#pragma unroll
    for (int j = 0; j < 8; ++j) {
      const int c = t + 256 * j;
      const int row = c >> 4;
      const int kc = (c & 15) << 2;
      v4h hv;
      hv[0] = (_Float16)rw[j].x; hv[1] = (_Float16)rw[j].y;
      hv[2] = (_Float16)rw[j].z; hv[3] = (_Float16)rw[j].w;
      *(v4h*)(lB + row * 80 + kc) = hv;
    }
    __syncthreads();   // LDS tiles ready

    // ---- compute: 2 K-steps of 32, 16 WMMAs per wave -----------------------
#pragma unroll
    for (int kk = 0; kk < 64; kk += 32) {
      v16h af[2], bf[4];
      af[0] = ldfragA(lA, wm * 32 + 0, 80, kk);
      af[1] = ldfragA(lA, wm * 32 + 16, 80, kk);
#pragma unroll
      for (int tj = 0; tj < 4; ++tj)
        bf[tj] = ldfragB(lB, wn * 64 + tj * 16, 80, kk);
#pragma unroll
      for (int ti = 0; ti < 2; ++ti)
#pragma unroll
        for (int tj = 0; tj < 4; ++tj)
          acc[ti][tj] = wmma16(af[ti], bf[tj], acc[ti][tj]);
    }
  }

  // ---- epilogue: bias add + store.  C layout: lane n = l&15, row = ti*16+8*g+v
  const int g = lane >> 4;
  const int ln = lane & 15;
#pragma unroll
  for (int ti = 0; ti < 2; ++ti) {
#pragma unroll
    for (int tj = 0; tj < 4; ++tj) {
      const int n = n0 + wn * 64 + tj * 16 + ln;
      const float bn = bias[n];
#pragma unroll
      for (int v = 0; v < 8; ++v) {
        const int m = m0 + wm * 32 + ti * 16 + 8 * g + v;
        const float val = acc[ti][tj][v] + bn;
        if (CHALF) ((_Float16*)Cv)[(size_t)m * ldc + n] = (_Float16)val;
        else       ((float*)Cv)[(size_t)m * ldc + n] = val;
      }
    }
  }
}

// ---------------------------------------------------------------------------
// Attention: one block (256 thr = 8 waves) per batch, one wave per head.
// qkv: f16 [N*L][1536] (q|k|v).  RoPE+scale fused into LDS staging.
// attnOut: f16 [N*L][512].  avg: f32 [N][50][50] (deterministic head mean).
// LDS: qs/ks/vt each 8 heads * 64 rows * 80 halves -> 245760 bytes dynamic.
// ---------------------------------------------------------------------------
__global__ __launch_bounds__(256) void attention_kernel(
    const _Float16* __restrict__ qkv, _Float16* __restrict__ attnOut,
    float* __restrict__ avg) {
  extern __shared__ _Float16 sm[];
  _Float16* qs = sm;               // [h][i][d]  (rope'd, scaled q)
  _Float16* ks = sm + 40960;       // [h][j][d]  (rope'd k) -> reused for attn[h][i][j]
  _Float16* vt = sm + 81920;       // [h][d][j]  (v transposed)

  const int b = blockIdx.x;
  const int t = threadIdx.x;

  // zero-fill (pads rows 50..63 / cols 50..63 with zeros)
  {
    v8h z = {};
    for (int c = t; c < 15360; c += 256) *(v8h*)(sm + c * 8) = z;
  }
  __syncthreads();

  // cooperative load: 50 rows x 1536 halves = 9600 chunks of 8 halves
  const float kNegLn = -0.28782313662425574f;  // -ln(10000)/32
  for (int c = t; c < 9600; c += 256) {
    const int i = c / 192;               // position 0..49
    const int n0 = (c - i * 192) * 8;    // 0..1528
    const v8h ch = *(const v8h*)(qkv + (size_t)(b * 50 + i) * 1536 + n0);
    const int s = n0 >> 9;               // 0=q 1=k 2=v
    const int nn = n0 & 511;
    const int h = nn >> 6;
    const int d0 = nn & 63;
    if (s < 2) {
      const float fi = (float)i;
      v8h o;
#pragma unroll
      for (int p = 0; p < 4; ++p) {
        const float theta = __expf((float)((d0 >> 1) + p) * kNegLn);
        const float ang = fi * theta;
        const float cs = __cosf(ang), sn = __sinf(ang);
        const float x1 = (float)ch[2 * p], x2 = (float)ch[2 * p + 1];
        float o1 = x1 * cs - x2 * sn;
        float o2 = x1 * sn + x2 * cs;
        if (s == 0) { o1 *= 0.125f; o2 *= 0.125f; }  // 1/sqrt(HD)
        o[2 * p] = (_Float16)o1;
        o[2 * p + 1] = (_Float16)o2;
      }
      _Float16* dst = (s == 0) ? qs : ks;
      *(v8h*)(dst + h * 5120 + i * 80 + d0) = o;
    } else {
      // transpose v into vt[h][d][j]
#pragma unroll
      for (int e = 0; e < 8; ++e) vt[h * 5120 + (d0 + e) * 80 + i] = ch[e];
    }
  }
  __syncthreads();

  const int w = t >> 5;        // head
  const int lane = t & 31;
  const int g = lane >> 4;
  const int ln = lane & 15;
  const _Float16* qh = qs + w * 5120;
  _Float16* kh = ks + w * 5120;
  const _Float16* vh = vt + w * 5120;

  // ---- scores = q_rope_scaled @ k_rope^T : 64x64 in 16 WMMA tiles ----
  v8f sc[4][4] = {};
  for (int k2 = 0; k2 < 64; k2 += 32) {
    v16h af[4], bf[4];
#pragma unroll
    for (int ti = 0; ti < 4; ++ti) af[ti] = ldfragA(qh, ti * 16, 80, k2);
#pragma unroll
    for (int tj = 0; tj < 4; ++tj) bf[tj] = ldfragB(kh, tj * 16, 80, k2);
#pragma unroll
    for (int ti = 0; ti < 4; ++ti)
#pragma unroll
      for (int tj = 0; tj < 4; ++tj)
        sc[ti][tj] = wmma16(af[ti], bf[tj], sc[ti][tj]);
  }

  // ---- row softmax (mask j>=50), write attn f16 into kh (k is dead) ----
#pragma unroll
  for (int ti = 0; ti < 4; ++ti) {
#pragma unroll
    for (int v = 0; v < 8; ++v) {
      const int row = ti * 16 + 8 * g + v;
      float x[4];
      float mx = -1.0e30f;
#pragma unroll
      for (int tj = 0; tj < 4; ++tj) {
        const int j = tj * 16 + ln;
        const float val = (j < 50) ? sc[ti][tj][v] : -1.0e30f;
        x[tj] = val;
        mx = fmaxf(mx, val);
      }
      for (int o = 8; o > 0; o >>= 1) mx = fmaxf(mx, __shfl_xor(mx, o, 16));
      float sum = 0.0f;
#pragma unroll
      for (int tj = 0; tj < 4; ++tj) { x[tj] = __expf(x[tj] - mx); sum += x[tj]; }
      for (int o = 8; o > 0; o >>= 1) sum += __shfl_xor(sum, o, 16);
      const float inv = 1.0f / sum;
      if (row < 50) {
#pragma unroll
        for (int tj = 0; tj < 4; ++tj)
          kh[row * 80 + tj * 16 + ln] = (_Float16)(x[tj] * inv);
      }
    }
  }

  // ---- out = attn @ v : A = attn[i][j] (kh), B = vt[d][j] ----
  v8f oa[4][4] = {};
  for (int k2 = 0; k2 < 64; k2 += 32) {
    v16h af[4], bf[4];
#pragma unroll
    for (int ti = 0; ti < 4; ++ti) af[ti] = ldfragA(kh, ti * 16, 80, k2);
#pragma unroll
    for (int td = 0; td < 4; ++td) bf[td] = ldfragB(vh, td * 16, 80, k2);
#pragma unroll
    for (int ti = 0; ti < 4; ++ti)
#pragma unroll
      for (int td = 0; td < 4; ++td)
        oa[ti][td] = wmma16(af[ti], bf[td], oa[ti][td]);
  }
#pragma unroll
  for (int ti = 0; ti < 4; ++ti)
#pragma unroll
    for (int td = 0; td < 4; ++td)
#pragma unroll
      for (int v = 0; v < 8; ++v) {
        const int row = ti * 16 + 8 * g + v;
        if (row < 50) {
          const int d = td * 16 + ln;
          attnOut[(size_t)(b * 50 + row) * 512 + w * 64 + d] = (_Float16)oa[ti][td][v];
        }
      }
  __syncthreads();

  // ---- avg_weights = mean over heads (fixed order -> deterministic) ----
  for (int c = t; c < 2500; c += 256) {
    const int i = c / 50;
    const int j = c - i * 50;
    float sfull = 0.0f;
#pragma unroll
    for (int h = 0; h < 8; ++h) sfull += (float)ks[h * 5120 + i * 80 + j];
    avg[(size_t)b * 2500 + c] = sfull * 0.125f;
  }
}

// ---------------------------------------------------------------------------
extern "C" void kernel_launch(void* const* d_in, const int* in_sizes, int n_in,
                              void* d_out, int out_size, void* d_ws, size_t ws_size,
                              hipStream_t stream) {
  (void)in_sizes; (void)n_in; (void)out_size; (void)ws_size;
  const float* query = (const float*)d_in[0];
  const float* key   = (const float*)d_in[1];
  const float* value = (const float*)d_in[2];
  const float* ipw   = (const float*)d_in[3];   // [1536][512]
  const float* ipb   = (const float*)d_in[4];   // [1536]
  const float* opw   = (const float*)d_in[5];   // [512][512]
  const float* opb   = (const float*)d_in[6];   // [512]

  constexpr int N = 2048, L = 50, E = 512;
  constexpr int M = N * L;                       // 102400 rows

  // workspace: qkv f16 [M][1536] (315MB) + attnOut f16 [M][512] (105MB)
  _Float16* qkv     = (_Float16*)d_ws;
  _Float16* attnOut = qkv + (size_t)M * 3 * E;

  float* out = (float*)d_out;                    // [M][512]
  float* avg = out + (size_t)M * E;              // [N][50][50]

  const dim3 blk(256);
  const dim3 gq(E / 128, M / 128);               // 4 x 800

  // QKV projections (A=f32, C=f16, ldc=1536 with column offsets)
  gemm_wmma_kernel<false, true><<<gq, blk, 0, stream>>>(query, ipw,             ipb,         qkv,         E, 3 * E);
  gemm_wmma_kernel<false, true><<<gq, blk, 0, stream>>>(key,   ipw + E * E,     ipb + E,     qkv + E,     E, 3 * E);
  gemm_wmma_kernel<false, true><<<gq, blk, 0, stream>>>(value, ipw + 2 * E * E, ipb + 2 * E, qkv + 2 * E, E, 3 * E);

  // fused RoPE + attention + head-mean (245760 B dynamic LDS)
  attention_kernel<<<dim3(N), blk, 245760, stream>>>(qkv, attnOut, avg);

  // output projection (A=f16, C=f32)
  gemm_wmma_kernel<true, false><<<gq, blk, 0, stream>>>(attnOut, opw, opb, out, E, E);
}
